// DualChannelGAT_29703993819792
// MI455X (gfx1250) — matrix operand order repaired and checked by reading, not compiled, not run on hardware
//
#include <hip/hip_runtime.h>
#include <hip/hip_bf16.h>
#include <cstdint>
#include <cstddef>

// ---------------------------------------------------------------------------
// Types for CDNA5 WMMA (gfx1250, wave32)
// ---------------------------------------------------------------------------
typedef __bf16 bf16_t;
typedef __attribute__((ext_vector_type(16))) __bf16 v16bf;
typedef __attribute__((ext_vector_type(8)))  __bf16 v8bf;
typedef __attribute__((ext_vector_type(8)))  float  v8f;

#define B_  8
#define N_  1024
#define F_  512
#define H_  8
#define O_  64
#define NW_ 32           // N_/32 packed mask words per row
#define NEGV (-9e15f)
#define LHS 40           // LDS row stride (bf16) for transposed h tile (16B aligned, bank-spread)

// CDNA5 async global->LDS path, gated so either toolchain compiles.
#if defined(__AMDGCN__) &&                                                   \
    __has_builtin(__builtin_amdgcn_global_load_async_to_lds_b128) &&         \
    __has_builtin(__builtin_amdgcn_s_wait_asynccnt)
#define HAVE_ASYNC_LDS 1
#else
#define HAVE_ASYNC_LDS 0
#endif

#if HAVE_ASYNC_LDS
// Builtin expects: (v4i32 AS1*, v4i32 AS3*, imm offset, imm cpol)
typedef int async_v4i __attribute__((vector_size(16)));
typedef __attribute__((address_space(1))) async_v4i* async_gp;
typedef __attribute__((address_space(3))) async_v4i* async_lp;
#endif

// ---------------------------------------------------------------------------
// wave32 reductions
// ---------------------------------------------------------------------------
__device__ __forceinline__ float warp_max32(float v) {
#pragma unroll
  for (int off = 16; off; off >>= 1) v = fmaxf(v, __shfl_xor(v, off, 32));
  return v;
}
__device__ __forceinline__ float warp_sum32(float v) {
#pragma unroll
  for (int off = 16; off; off >>= 1) v += __shfl_xor(v, off, 32);
  return v;
}

// 16B-vectorized fragment loads (ISA 7.12.2 layouts).
// A 16x32 frag: lane half=0 -> K {0..7,16..23}; half=1 -> {8..15,24..31}
__device__ __forceinline__ v16bf load_a_frag(const bf16_t* p) {
  v8bf lo = *(const v8bf*)p;
  v8bf hi = *(const v8bf*)(p + 16);
  return __builtin_shufflevector(lo, hi, 0, 1, 2, 3, 4, 5, 6, 7, 8, 9, 10, 11,
                                 12, 13, 14, 15);
}
// B 32x16 frag: lane half=0 -> K {0..7,16..23}; half=1 -> {8..15,24..31}
__device__ __forceinline__ v16bf load_b_frag(const bf16_t* p) {
  v8bf lo = *(const v8bf*)p;
  v8bf hi = *(const v8bf*)(p + 16);
  return __builtin_shufflevector(lo, hi, 0, 1, 2, 3, 4, 5, 6, 7, 8, 9, 10, 11,
                                 12, 13, 14, 15);
}

// ---------------------------------------------------------------------------
// WMMA GEMM:  C[M,Nc] = act(A[M,K](bf16) @ W^T + bias) (+ Res)
// W is [Nc,K] bf16 row-major ("[out,in]"). One wave -> 32x64 C macro-tile
// (2 A-frags x 4 B-frags -> 8 v_wmma per 32-K step).
// M % 32 == 0, Nc % 64 == 0, K % 32 == 0.
// ACT: 0 none, 1 relu, 2 elu, 3 sigmoid, 4 leaky(0.2)
// ---------------------------------------------------------------------------
template <int ACT, bool RES>
__global__ __launch_bounds__(256) void gemm_wmma(
    const bf16_t* __restrict__ A, const bf16_t* __restrict__ Bm,
    const float* __restrict__ bias, const float* __restrict__ Res,
    float* __restrict__ C, int M, int Nc, int K) {
  int wid = blockIdx.x * 8 + (threadIdx.x >> 5);
  int tiles_n = Nc >> 6;
  int tm = wid / tiles_n;
  int tn = wid - tm * tiles_n;
  if (tm >= (M >> 5)) return;  // wave-uniform exit (EXEC stays all-1)
  int lane = threadIdx.x & 31;
  int half = lane >> 4, l16 = lane & 15;

  const bf16_t* arow0 = A + (size_t)((tm << 5) + l16) * K + half * 8;
  const bf16_t* arow1 = arow0 + (size_t)16 * K;
  const bf16_t* brow0 = Bm + (size_t)((tn << 6) + l16) * K + half * 8;
  const size_t bstep = (size_t)16 * K;

  v8f zero = {0.f, 0.f, 0.f, 0.f, 0.f, 0.f, 0.f, 0.f};
  v8f acc[8];
#pragma unroll
  for (int i = 0; i < 8; ++i) acc[i] = zero;

  for (int k0 = 0; k0 < K; k0 += 32) {
    __builtin_prefetch(arow0 + k0 + 128, 0, 1);
    __builtin_prefetch(brow0 + k0 + 128, 0, 1);
    v16bf af0 = load_a_frag(arow0 + k0);
    v16bf af1 = load_a_frag(arow1 + k0);
    v16bf bf0 = load_b_frag(brow0 + k0);
    v16bf bf1 = load_b_frag(brow0 + bstep + k0);
    v16bf bf2 = load_b_frag(brow0 + 2 * bstep + k0);
    v16bf bf3 = load_b_frag(brow0 + 3 * bstep + k0);
    acc[0] = __builtin_amdgcn_wmma_f32_16x16x32_bf16(false, af0, false, bf0, (short)0, acc[0], false, false);
    acc[1] = __builtin_amdgcn_wmma_f32_16x16x32_bf16(false, af0, false, bf1, (short)0, acc[1], false, false);
    acc[2] = __builtin_amdgcn_wmma_f32_16x16x32_bf16(false, af0, false, bf2, (short)0, acc[2], false, false);
    acc[3] = __builtin_amdgcn_wmma_f32_16x16x32_bf16(false, af0, false, bf3, (short)0, acc[3], false, false);
    acc[4] = __builtin_amdgcn_wmma_f32_16x16x32_bf16(false, af1, false, bf0, (short)0, acc[4], false, false);
    acc[5] = __builtin_amdgcn_wmma_f32_16x16x32_bf16(false, af1, false, bf1, (short)0, acc[5], false, false);
    acc[6] = __builtin_amdgcn_wmma_f32_16x16x32_bf16(false, af1, false, bf2, (short)0, acc[6], false, false);
    acc[7] = __builtin_amdgcn_wmma_f32_16x16x32_bf16(false, af1, false, bf3, (short)0, acc[7], false, false);
  }
  // C layout per tile: VGPR r, lane -> M = r + 8*half, N = l16
#pragma unroll
  for (int mi = 0; mi < 2; ++mi) {
#pragma unroll
    for (int ni = 0; ni < 4; ++ni) {
      const v8f& a = acc[mi * 4 + ni];
      int n = (tn << 6) + ni * 16 + l16;
      float bv = bias ? bias[n] : 0.f;
#pragma unroll
      for (int r = 0; r < 8; ++r) {
        int m = (tm << 5) + mi * 16 + r + half * 8;
        float v = a[r] + bv;
        if (ACT == 1)      v = v > 0.f ? v : 0.f;
        else if (ACT == 2) v = v > 0.f ? v : (__expf(v) - 1.f);
        else if (ACT == 3) v = 1.f / (1.f + __expf(-v));
        else if (ACT == 4) v = v > 0.f ? v : 0.2f * v;
        if (RES) v += Res[(size_t)m * Nc + n];
        C[(size_t)m * Nc + n] = v;
      }
    }
  }
}

// ---------------------------------------------------------------------------
// f32 -> bf16 convert
// ---------------------------------------------------------------------------
__global__ void f32_to_bf16(const float* __restrict__ X, bf16_t* __restrict__ Y,
                            size_t n) {
  size_t i = (size_t)blockIdx.x * blockDim.x + threadIdx.x;
  if (i < n) Y[i] = (bf16_t)X[i];
}

// Regroup GAT head weights [L,H,F,O] (f32) -> [L, H*O, F] (bf16, "[out,in]")
__global__ void regroup_W(const float* __restrict__ W, bf16_t* __restrict__ Y) {
  size_t i = (size_t)blockIdx.x * blockDim.x + threadIdx.x;
  size_t total = (size_t)2 * H_ * O_ * F_;
  if (i >= total) return;
  int per = H_ * O_ * F_;
  int l = (int)(i / per);
  int r = (int)(i - (size_t)l * per);
  int n = r / F_;
  int k = r - n * F_;
  int hd = n >> 6, o = n & 63;
  Y[i] = (bf16_t)W[(((size_t)l * H_ + hd) * F_ + k) * O_ + o];
}

// h [b*N+m][h*64+o] (f32) -> transposed bf16 [ (b*H+h)*64 + o ][ m ]
__global__ void h_to_bf16_T(const float* __restrict__ Hf, bf16_t* __restrict__ Y,
                            int H, int rowlen) {
  size_t i = (size_t)blockIdx.x * blockDim.x + threadIdx.x;
  size_t total = (size_t)B_ * H * 64 * N_;
  if (i >= total) return;
  int m = (int)(i & (N_ - 1));
  int bho = (int)(i >> 10);
  int b = bho / (H * 64);
  int ho = bho - b * H * 64;
  int h = ho >> 6, o = ho & 63;
  Y[i] = (bf16_t)Hf[(size_t)(b * N_ + m) * rowlen + h * 64 + o];
}

// ---------------------------------------------------------------------------
// Pack adjacency to bitmask words: Pk[b*N*32 + n*32 + w] bits over m
// ---------------------------------------------------------------------------
template <bool COMB>
__global__ void pack_adj(const int* __restrict__ A1, const int* __restrict__ A2,
                         unsigned* __restrict__ Pk) {
  int w = blockIdx.x * blockDim.x + threadIdx.x;
  if (w >= B_ * N_ * NW_) return;
  size_t base = (size_t)w * 32;
  unsigned bits = 0;
#pragma unroll 4
  for (int j = 0; j < 32; ++j) {
    int v = A1[base + j];
    if (COMB) v += A2[base + j];
    bits |= (v > 0 ? 1u : 0u) << j;
  }
  Pk[w] = bits;
}

// ---------------------------------------------------------------------------
// s = h . a_src, d = h . a_dst per (b,h,n). Hf row layout [b*N+n][H*64]
// ---------------------------------------------------------------------------
__global__ void compute_sd(const float* __restrict__ Hf,
                           const float* __restrict__ Asrc,
                           const float* __restrict__ Adst,
                           float* __restrict__ S, float* __restrict__ Dv,
                           int H, int rowlen) {
  int i = blockIdx.x * blockDim.x + threadIdx.x;
  int total = B_ * H * N_;
  if (i >= total) return;
  int b = i / (H * N_);
  int r = i - b * H * N_;
  int h = r / N_;
  int n = r - h * N_;
  const float* hr = Hf + (size_t)(b * N_ + n) * rowlen + h * O_;
  const float* as = Asrc + h * O_;
  const float* ad = Adst + h * O_;
  float s = 0.f, d = 0.f;
#pragma unroll 8
  for (int o = 0; o < O_; ++o) { float v = hr[o]; s += v * as[o]; d += v * ad[o]; }
  S[i] = s; Dv[i] = d;
}

// ---------------------------------------------------------------------------
// Masked-softmax row stats: one wave per (b,h,n) query row.
// ---------------------------------------------------------------------------
__global__ __launch_bounds__(256) void att_rowstats(
    const float* __restrict__ S, const float* __restrict__ D,
    const unsigned* __restrict__ Pk, float* __restrict__ Rmax,
    float* __restrict__ Rinv, int H) {
  int w = blockIdx.x * 8 + (threadIdx.x >> 5);
  int lane = threadIdx.x & 31;
  if (w >= B_ * H * N_) return;
  int b = w / (H * N_);
  int n = w % N_;
  float sv = S[w];
  const float* drow = D + (size_t)(w / N_) * N_;
  const unsigned* pkrow = Pk + (size_t)(b * N_ + n) * NW_;
  float vmax = NEGV;
  for (int i = 0; i < NW_; ++i) {
    unsigned bits = pkrow[i];
    float e = sv + drow[i * 32 + lane];
    e = e > 0.f ? e : 0.2f * e;
    float v = ((bits >> lane) & 1u) ? e : NEGV;
    vmax = fmaxf(vmax, v);
  }
  vmax = warp_max32(vmax);
  float ssum = 0.f;
  for (int i = 0; i < NW_; ++i) {
    unsigned bits = pkrow[i];
    float e = sv + drow[i * 32 + lane];
    e = e > 0.f ? e : 0.2f * e;
    float v = ((bits >> lane) & 1u) ? e : NEGV;
    ssum += __expf(v - vmax);
  }
  ssum = warp_sum32(ssum);
  if (lane == 0) { Rmax[w] = vmax; Rinv[w] = 1.f / ssum; }
}

// ---------------------------------------------------------------------------
// Fused attention apply with WMMA.
// HbT: transposed bf16 h, [ (b*H+h)*64 + o ][ m ] (m contiguous).
// Tile (64 o-rows x 32 m) staged to LDS (async global->LDS when available),
// B-fragments read as two ds_load_b128. A fragment = softmax probabilities.
// ---------------------------------------------------------------------------
template <bool ELU>
__global__ __launch_bounds__(256) void att_apply(
    const bf16_t* __restrict__ HbT, const float* __restrict__ S,
    const float* __restrict__ D, const float* __restrict__ Rmax,
    const float* __restrict__ Rinv, const unsigned* __restrict__ Pk,
    float* __restrict__ Out, int H, int rowlenOut) {
  __shared__ bf16_t lhT[64 * LHS];   // [o][m] with padded row stride
  __shared__ float  ldd[32];
  int b = blockIdx.z, h = blockIdx.y;
  int wave = threadIdx.x >> 5, lane = threadIdx.x & 31;
  int half = lane >> 4, l16 = lane & 15;
  int n0 = blockIdx.x * 128 + wave * 16;
  int qn = n0 + l16;
  size_t srow = (size_t)(b * H + h) * N_;
  float sv = S[srow + qn];
  float rm = Rmax[srow + qn];
  const unsigned* pkrow = Pk + (size_t)(b * N_ + qn) * NW_;
  const bf16_t* hTbase = HbT + (size_t)(b * H + h) * 64 * N_;
  int so = threadIdx.x >> 2;        // staging o row (0..63)
  int sq = threadIdx.x & 3;         // staging quarter (8 bf16 = 16B)

  v8f zero = {0.f, 0.f, 0.f, 0.f, 0.f, 0.f, 0.f, 0.f};
  v8f acc[4] = {zero, zero, zero, zero};

  for (int m0 = 0; m0 < N_; m0 += 32) {
    // ---- stage h^T tile [64 x 32] into LDS ----
    {
      const bf16_t* gsrc = hTbase + (size_t)so * N_ + m0 + sq * 8;
      bf16_t* ldst = &lhT[so * LHS + sq * 8];
#if HAVE_ASYNC_LDS
      __builtin_amdgcn_global_load_async_to_lds_b128((async_gp)gsrc,
                                                     (async_lp)ldst, 0, 0);
#else
      *(v8bf*)ldst = *(const v8bf*)gsrc;
#endif
    }
    if (threadIdx.x < 32) ldd[threadIdx.x] = D[srow + m0 + threadIdx.x];
#if HAVE_ASYNC_LDS
    __builtin_amdgcn_s_wait_asynccnt(0);
#endif
    __syncthreads();

    // ---- A fragment: masked softmax probabilities (A K-layout) ----
    unsigned wbits = pkrow[m0 >> 5];
    v16bf av;
#pragma unroll
    for (int j = 0; j < 16; ++j) {
      int ml = (j < 8) ? (half * 8 + j) : (16 + half * 8 + (j - 8));
      float e = sv + ldd[ml];
      e = e > 0.f ? e : 0.2f * e;
      float p = ((wbits >> ml) & 1u) ? __expf(e - rm) : 0.f;
      av[j] = (bf16_t)p;
    }
    // ---- 4 WMMAs over output cols, B frags via 2x ds_load_b128 ----
#pragma unroll
    for (int oc = 0; oc < 4; ++oc) {
      const bf16_t* bp = &lhT[(oc * 16 + l16) * LHS + half * 8];
      v16bf bv = load_b_frag(bp);
      acc[oc] = __builtin_amdgcn_wmma_f32_16x16x32_bf16(
          false, av, false, bv, (short)0, acc[oc], false, false);
    }
    __syncthreads();
  }
  // ---- store: row scale 1/Z, optional ELU, into [b*N+q][h*64 + o] ----
#pragma unroll
  for (int oc = 0; oc < 4; ++oc) {
#pragma unroll
    for (int r = 0; r < 8; ++r) {
      int qrow = n0 + r + half * 8;
      float v = acc[oc][r] * Rinv[srow + qrow];
      if (ELU) v = v > 0.f ? v : (__expf(v) - 1.f);
      Out[(size_t)(b * N_ + qrow) * rowlenOut + h * 64 + oc * 16 + l16] = v;
    }
  }
}

// ---------------------------------------------------------------------------
// Small helper kernels
// ---------------------------------------------------------------------------
__global__ void mean_nodes(const float* __restrict__ X, float* __restrict__ Y, int F) {
  int i = blockIdx.x * blockDim.x + threadIdx.x;
  if (i >= B_ * F) return;
  int b = i / F, f = i - b * F;
  float s = 0.f;
  for (int n = 0; n < N_; ++n) s += X[(size_t)(b * N_ + n) * F + f];
  Y[i] = s * (1.f / N_);
}

__global__ __launch_bounds__(256) void se_mlp(const float* __restrict__ P,
                                              const float* __restrict__ W1,
                                              const float* __restrict__ W2,
                                              float* __restrict__ Y) {
  __shared__ float ps[F_];
  __shared__ float y1[32];
  int b = blockIdx.x, t = threadIdx.x;
  for (int i = t; i < F_; i += 256) ps[i] = P[b * F_ + i];
  __syncthreads();
  if (t < 32) {
    float s = 0.f;
    const float* w = W1 + t * F_;
    for (int f = 0; f < F_; ++f) s += ps[f] * w[f];
    y1[t] = s > 0.f ? s : 0.f;
  }
  __syncthreads();
  for (int f = t; f < F_; f += 256) {
    float s = 0.f;
    const float* w = W2 + f * 32;
#pragma unroll
    for (int k = 0; k < 32; ++k) s += y1[k] * w[k];
    Y[b * F_ + f] = 1.f / (1.f + __expf(-s));
  }
}

__global__ void se_scale(float* __restrict__ X, const float* __restrict__ Y) {
  size_t i = (size_t)blockIdx.x * blockDim.x + threadIdx.x;
  if (i >= (size_t)B_ * N_ * F_) return;
  int b = (int)(i / ((size_t)N_ * F_));
  int f = (int)(i % F_);
  X[i] *= Y[b * F_ + f];
}

__global__ void concat_bf16(const float* __restrict__ A, const float* __restrict__ Bx,
                            bf16_t* __restrict__ Y) {
  size_t i = (size_t)blockIdx.x * blockDim.x + threadIdx.x;
  if (i >= (size_t)B_ * N_ * 1024) return;
  size_t row = i >> 10;
  int col = (int)(i & 1023);
  float v = (col < F_) ? A[row * F_ + col] : Bx[row * F_ + (col - F_)];
  Y[i] = (bf16_t)v;
}

__global__ void gate_combine(const float* __restrict__ G1, const float* __restrict__ G2,
                             const float* __restrict__ F1v, const float* __restrict__ F2v,
                             bf16_t* __restrict__ Y) {
  size_t i = (size_t)blockIdx.x * blockDim.x + threadIdx.x;
  if (i >= (size_t)B_ * N_ * 1024) return;
  size_t row = i >> 10;
  int col = (int)(i & 1023);
  float v;
  if (col < F_) {
    float g = G1[row * F_ + col];
    v = g * F1v[row * F_ + col] + (1.f - g) * F2v[row * F_ + col];
  } else {
    int c = col - F_;
    float g = G2[row * F_ + c];
    v = g * F2v[row * F_ + c] + (1.f - g) * F1v[row * F_ + c];
  }
  Y[i] = (bf16_t)v;
}

__global__ void al_linear(const float* __restrict__ P, const float* __restrict__ W,
                          const float* __restrict__ Bs, float* __restrict__ Y) {
  int i = blockIdx.x * blockDim.x + threadIdx.x;
  if (i >= B_ * O_) return;
  int b = i >> 6, f = i & 63;
  float s = Bs[f];
  const float* w = W + f * O_;
#pragma unroll 8
  for (int k = 0; k < O_; ++k) s += P[b * O_ + k] * w[k];
  Y[i] = s;
}

__global__ __launch_bounds__(128) void final_head(
    const float* __restrict__ X, const float* __restrict__ G,
    const float* __restrict__ Bb, const float* __restrict__ CW,
    const float* __restrict__ CB, float* __restrict__ Out) {
  __shared__ float fs[B_ * O_];
  int t = threadIdx.x;
  if (t < O_) {
    float mu = 0.f;
    for (int b = 0; b < B_; ++b) mu += X[b * O_ + t];
    mu *= (1.f / B_);
    float var = 0.f;
    for (int b = 0; b < B_; ++b) { float d = X[b * O_ + t] - mu; var += d * d; }
    var *= (1.f / B_);
    float inv = rsqrtf(var + 1e-5f);
    for (int b = 0; b < B_; ++b) {
      float v = (X[b * O_ + t] - mu) * inv * G[t] + Bb[t];
      v = v > 0.f ? v : 0.2f * v;
      fs[b * O_ + t] = v;
      Out[16 + b * O_ + t] = v;     // feats
    }
  }
  __syncthreads();
  if (t < 16) {
    int b = t >> 1, c = t & 1;
    float s = CB[c];
    const float* w = CW + c * O_;
#pragma unroll 8
    for (int k = 0; k < O_; ++k) s += fs[b * O_ + k] * w[k];
    Out[b * 2 + c] = s;             // logits
  }
}

// ---------------------------------------------------------------------------
// Host orchestration
// ---------------------------------------------------------------------------
#define GEMM(ACT, RES, Ap, Bp, biasp, Resp, Cp, Mv, Nv, Kv)                   \
  gemm_wmma<ACT, RES>                                                         \
      <<<dim3((unsigned)((((Mv) / 32) * ((Nv) / 64) + 7) / 8)), 256, 0,       \
         stream>>>((Ap), (Bp), (biasp), (Resp), (Cp), (Mv), (Nv), (Kv))

extern "C" void kernel_launch(void* const* d_in, const int* in_sizes, int n_in,
                              void* d_out, int out_size, void* d_ws,
                              size_t ws_size, hipStream_t stream) {
  (void)in_sizes; (void)n_in; (void)out_size; (void)ws_size;
  const float* fea    = (const float*)d_in[0];
  const int*   adj1   = (const int*)d_in[1];
  const int*   adj2   = (const int*)d_in[2];
  const float* W1     = (const float*)d_in[3];
  const float* asrc1  = (const float*)d_in[4];
  const float* adst1  = (const float*)d_in[5];
  const float* red1w  = (const float*)d_in[6];
  const float* red1b  = (const float*)d_in[7];
  const float* W2     = (const float*)d_in[8];
  const float* asrc2  = (const float*)d_in[9];
  const float* adst2  = (const float*)d_in[10];
  const float* red2w  = (const float*)d_in[11];
  const float* red2b  = (const float*)d_in[12];
  const float* se1w1  = (const float*)d_in[13];
  const float* se1w2  = (const float*)d_in[14];
  const float* se2w1  = (const float*)d_in[15];
  const float* se2w2  = (const float*)d_in[16];
  const float* gitw   = (const float*)d_in[17];
  const float* gitb   = (const float*)d_in[18];
  const float* gsw1   = (const float*)d_in[19];
  const float* gsb1   = (const float*)d_in[20];
  const float* gsw2   = (const float*)d_in[21];
  const float* gsb2   = (const float*)d_in[22];
  const float* gow1   = (const float*)d_in[23];
  const float* gob1   = (const float*)d_in[24];
  const float* gow2   = (const float*)d_in[25];
  const float* gob2   = (const float*)d_in[26];
  const float* frw1   = (const float*)d_in[27];
  const float* frb1   = (const float*)d_in[28];
  const float* frw2   = (const float*)d_in[29];
  const float* frb2   = (const float*)d_in[30];
  const float* oaw    = (const float*)d_in[31];
  const float* oasrc  = (const float*)d_in[32];
  const float* oadst  = (const float*)d_in[33];
  const float* alw    = (const float*)d_in[34];
  const float* alb    = (const float*)d_in[35];
  const float* bng    = (const float*)d_in[36];
  const float* bnb    = (const float*)d_in[37];
  const float* clw    = (const float*)d_in[38];
  const float* clb    = (const float*)d_in[39];

  const size_t R = (size_t)B_ * N_;   // 8192 rows

  // ---- workspace layout ----
  char* ws = (char*)d_ws;
  size_t off = 0;
  auto salloc = [&](size_t bytes) -> char* {
    char* p = ws + off;
    off += (bytes + 255) & ~(size_t)255;
    return p;
  };
  float*   x1  = (float*)salloc(R * F_ * 4);
  float*   x2  = (float*)salloc(R * F_ * 4);
  float*   xa  = (float*)salloc(R * F_ * 4);
  float*   hf  = (float*)salloc(R * F_ * 4);
  bf16_t*  cb  = (bf16_t*)salloc(R * 1024 * 2);
  bf16_t*  xb  = (bf16_t*)salloc(R * F_ * 2);
  bf16_t*  hb  = (bf16_t*)salloc(R * F_ * 2);   // holds transposed h (bf16)
  unsigned* pk = (unsigned*)salloc((size_t)B_ * N_ * NW_ * 4);
  float*   s_  = (float*)salloc((size_t)B_ * H_ * N_ * 4);
  float*   dv  = (float*)salloc((size_t)B_ * H_ * N_ * 4);
  float*   rm  = (float*)salloc((size_t)B_ * H_ * N_ * 4);
  float*   ri  = (float*)salloc((size_t)B_ * H_ * N_ * 4);
  // bf16 weights (all in [out,in] layout)
  bf16_t* w1g   = (bf16_t*)salloc((size_t)2 * H_ * O_ * F_ * 2);
  bf16_t* w2g   = (bf16_t*)salloc((size_t)2 * H_ * O_ * F_ * 2);
  bf16_t* red1wb= (bf16_t*)salloc((size_t)F_ * F_ * 2);
  bf16_t* red2wb= (bf16_t*)salloc((size_t)F_ * F_ * 2);
  bf16_t* gitwb = (bf16_t*)salloc((size_t)F_ * F_ * 2);
  bf16_t* gsw1b = (bf16_t*)salloc((size_t)F_ * 1024 * 2);
  bf16_t* gsw2b = (bf16_t*)salloc((size_t)F_ * F_ * 2);
  bf16_t* gow1b = (bf16_t*)salloc((size_t)F_ * 1024 * 2);
  bf16_t* gow2b = (bf16_t*)salloc((size_t)F_ * F_ * 2);
  bf16_t* frw1b = (bf16_t*)salloc((size_t)F_ * 1024 * 2);
  bf16_t* frw2b = (bf16_t*)salloc((size_t)O_ * F_ * 2);
  bf16_t* oawb  = (bf16_t*)salloc((size_t)O_ * O_ * 2);
  float*  pool  = (float*)salloc((size_t)B_ * F_ * 4);
  float*  yv    = (float*)salloc((size_t)B_ * F_ * 4);
  float*  xal   = (float*)salloc((size_t)B_ * O_ * 4);

  auto cvt = [&](const float* src, bf16_t* dst, size_t n) {
    f32_to_bf16<<<dim3((unsigned)((n + 255) / 256)), 256, 0, stream>>>(src, dst, n);
  };

  // ---- weight conversions (bf16, [out,in]) ----
  regroup_W<<<dim3((unsigned)((2 * H_ * O_ * F_) / 256)), 256, 0, stream>>>(W1, w1g);
  regroup_W<<<dim3((unsigned)((2 * H_ * O_ * F_) / 256)), 256, 0, stream>>>(W2, w2g);
  cvt(red1w, red1wb, (size_t)F_ * F_);
  cvt(red2w, red2wb, (size_t)F_ * F_);
  cvt(gitw, gitwb, (size_t)F_ * F_);
  cvt(gsw1, gsw1b, (size_t)F_ * 1024);
  cvt(gsw2, gsw2b, (size_t)F_ * F_);
  cvt(gow1, gow1b, (size_t)F_ * 1024);
  cvt(gow2, gow2b, (size_t)F_ * F_);
  cvt(frw1, frw1b, (size_t)F_ * 1024);
  cvt(frw2, frw2b, (size_t)O_ * F_);
  cvt(oaw, oawb, (size_t)O_ * O_);

  // ---- two GAT channels ----
  struct Chan {
    const int* adj; const bf16_t* Wg; const float* asrc; const float* adst;
    const bf16_t* redwb; const float* redb; float* xout;
  } chans[2] = {
      {adj1, w1g, asrc1, adst1, red1wb, red1b, x1},
      {adj2, w2g, asrc2, adst2, red2wb, red2b, x2},
  };
  for (int c = 0; c < 2; ++c) {
    Chan& ch = chans[c];
    pack_adj<false><<<dim3((unsigned)(B_ * N_ * NW_ / 256)), 256, 0, stream>>>(
        ch.adj, ch.adj, pk);
    const float* xcur = fea;
    float* bufs[2] = {xa, ch.xout};
    for (int l = 0; l < 2; ++l) {
      cvt(xcur, xb, R * F_);
      // h (all heads) = x @ W[l]^T -> [R, H*64]
      GEMM(0, false, xb, ch.Wg + (size_t)l * H_ * O_ * F_, nullptr, nullptr,
           hf, (int)R, 512, 512);
      h_to_bf16_T<<<dim3((unsigned)(R * F_ / 256)), 256, 0, stream>>>(hf, hb, H_, F_);
      compute_sd<<<dim3((unsigned)(B_ * H_ * N_ / 256)), 256, 0, stream>>>(
          hf, ch.asrc + l * H_ * O_, ch.adst + l * H_ * O_, s_, dv, H_, F_);
      att_rowstats<<<dim3((unsigned)(B_ * H_ * N_ / 8)), 256, 0, stream>>>(
          s_, dv, pk, rm, ri, H_);
      att_apply<true><<<dim3(8, H_, B_), 256, 0, stream>>>(hb, s_, dv, rm, ri,
                                                           pk, hf, H_, F_);
      cvt(hf, xb, R * F_);  // hcat -> bf16
      // x = relu(hcat @ red_w^T + red_b) + x
      GEMM(1, true, xb, ch.redwb, ch.redb, xcur, bufs[l], (int)R, 512, 512);
      xcur = bufs[l];
    }
  }

  // ---- SE blocks ----
  mean_nodes<<<dim3((unsigned)((B_ * F_ + 255) / 256)), 256, 0, stream>>>(x1, pool, F_);
  se_mlp<<<dim3(B_), 256, 0, stream>>>(pool, se1w1, se1w2, yv);
  se_scale<<<dim3((unsigned)(R * F_ / 256)), 256, 0, stream>>>(x1, yv);
  mean_nodes<<<dim3((unsigned)((B_ * F_ + 255) / 256)), 256, 0, stream>>>(x2, pool, F_);
  se_mlp<<<dim3(B_), 256, 0, stream>>>(pool, se2w1, se2w2, yv);
  se_scale<<<dim3((unsigned)(R * F_ / 256)), 256, 0, stream>>>(x2, yv);

  // ---- gated interaction ----
  cvt(x2, xb, R * F_);
  GEMM(1, true, xb, gitwb, gitb, x1, xa, (int)R, 512, 512);   // f1 -> xa
  cvt(x1, hb, R * F_);
  GEMM(1, true, hb, gitwb, gitb, x2, hf, (int)R, 512, 512);   // f2 -> hf
  concat_bf16<<<dim3((unsigned)(R * 1024 / 256)), 256, 0, stream>>>(xa, hf, cb);
  GEMM(1, false, cb, gsw1b, gsb1, nullptr, x1, (int)R, 512, 1024);  // gs hidden
  cvt(x1, xb, R * F_);
  GEMM(3, false, xb, gsw2b, gsb2, nullptr, x2, (int)R, 512, 512);   // g1 -> x2
  GEMM(1, false, cb, gow1b, gob1, nullptr, x1, (int)R, 512, 1024);  // go hidden
  cvt(x1, xb, R * F_);
  GEMM(3, false, xb, gow2b, gob2, nullptr, x1, (int)R, 512, 512);   // g2 -> x1
  gate_combine<<<dim3((unsigned)(R * 1024 / 256)), 256, 0, stream>>>(x2, x1, xa, hf, cb);

  // ---- final reduce MLP ----
  GEMM(1, false, cb, frw1b, frb1, nullptr, x1, (int)R, 512, 1024);  // hidden
  cvt(x1, xb, R * F_);
  GEMM(0, false, xb, frw2b, frb2, nullptr, x2, (int)R, 64, 512);    // xo -> x2

  // ---- output GAT (single head, 64 dim, no elu) ----
  cvt(x2, xb, R * 64);
  GEMM(0, false, xb, oawb, nullptr, nullptr, hf, (int)R, 64, 64);   // h -> hf
  h_to_bf16_T<<<dim3((unsigned)(R * 64 / 256)), 256, 0, stream>>>(hf, hb, 1, 64);
  compute_sd<<<dim3((unsigned)(B_ * N_ / 256)), 256, 0, stream>>>(hf, oasrc, oadst,
                                                                  s_, dv, 1, 64);
  pack_adj<true><<<dim3((unsigned)(B_ * N_ * NW_ / 256)), 256, 0, stream>>>(adj1, adj2, pk);
  att_rowstats<<<dim3((unsigned)(B_ * N_ / 8)), 256, 0, stream>>>(s_, dv, pk, rm, ri, 1);
  att_apply<false><<<dim3(8, 1, B_), 256, 0, stream>>>(hb, s_, dv, rm, ri, pk, x1, 1, 64);

  // ---- head: pool, al linear, BN + leaky + classifier ----
  mean_nodes<<<dim3((unsigned)((B_ * 64 + 255) / 256)), 256, 0, stream>>>(x1, pool, 64);
  al_linear<<<dim3(2), 256, 0, stream>>>(pool, alw, alb, xal);
  final_head<<<dim3(1), 128, 0, stream>>>(xal, bng, bnb, clw, clb, (float*)d_out);
}
